// Multi3GruUser_37563783971449
// MI455X (gfx1250) — matrix operand order repaired and checked by reading, not compiled, not run on hardware
//
#include <hip/hip_runtime.h>
#include <hip/hip_bf16.h>

// CDNA5 / gfx1250, wave32. Matrix ops via v_wmma_f32_16x16x32_f16 (fp32 accum).

typedef __attribute__((ext_vector_type(16))) _Float16 v16h;
typedef __attribute__((ext_vector_type(8)))  float    v8f;

// Global (address_space(1)) pointer type: forces global_load_* (LOADcnt only)
// instead of flat_load_* (LOADcnt + DScnt) for the laundered weight streams.
typedef const unsigned int __attribute__((address_space(1)))* gwp_t;

union AF { v16h v; unsigned int u[8]; };
union HP { unsigned int u; _Float16 h[2]; };

__device__ __forceinline__ float selu_(float x) {
    const float a = 1.6732632423543772f, s = 1.0507009873554805f;
    return x > 0.f ? s * x : s * a * (expf(x) - 1.f);
}

// ---------------------------------------------------------------------------
// Prep: fp32 [2][768][256] row-major -> f16 swizzled into WMMA B-operand
// fragments: dst index = (((dir*48 + ntile)*8 + ktile)*32 + lane)*8 + v
// B layout used here (contiguous K, not ISA table order): lane half (lane>>4) covers 16 consecutive K;
// dword v holds K pair {ktile*32 + 2v + 16*half, +1}; n = lane&15.
// ---------------------------------------------------------------------------
__global__ __launch_bounds__(256) void swizzle_weights_kernel(
    const float* __restrict__ src, unsigned int* __restrict__ dst, int total)
{
    int idx = blockIdx.x * blockDim.x + threadIdx.x;
    if (idx >= total) return;
    int v    = idx & 7;
    int t1   = idx >> 3;
    int lane = t1 & 31;
    int t2   = t1 >> 5;
    int kt   = t2 & 7;
    int t3   = t2 >> 3;
    int nt   = t3 % 48;
    int dir  = t3 / 48;
    int half = lane >> 4;
    int n    = lane & 15;
    int g  = nt * 16 + n;                 // gate row (0..767)
    int k0 = kt * 32 + 2 * v + 16 * half; // even k
    const float* row = src + (size_t)(dir * 768 + g) * 256;
    HP p;
    p.h[0] = (_Float16)row[k0];
    p.h[1] = (_Float16)row[k0 + 1];
    dst[idx] = p.u;
}

// ---------------------------------------------------------------------------
// Bi-GRU step kernel: one block = 32 batch rows (2 WMMA M-tiles) x one
// direction (blockIdx.y). 8 waves; wave w owns hidden columns [32w, 32w+32)
// for r, z AND n gates (ntiles {2w,2w+1, 16+2w,17+2w, 32+2w,33+2w}) so the
// gate nonlinearity is register-local. h lives in per-lane registers with an
// f16 mirror in LDS. Each B fragment is reused by both M-tiles (4 WMMAs).
// Pointer-laundering keeps the weight streams inside the time loop
// (L2-resident streaming) instead of being LICM-hoisted + spilled, and the
// AS(1) cast makes them global_load_b128 (LOADcnt-only tracking).
// mode 0: x_t = embed[tokens[b*T+tt]]   mode 1: x_t = x0[...] + x1[...]
// ---------------------------------------------------------------------------
__global__ __launch_bounds__(256) void bigru16_wmma_kernel(
    const unsigned int* __restrict__ wih_sw,
    const unsigned int* __restrict__ whh_sw,
    const float* __restrict__ bih,    // [2][768]
    const float* __restrict__ bhh,    // [2][768]
    const int*   __restrict__ tokens, // mode 0
    const float* __restrict__ embed,  // mode 0: [V][256]
    const float* __restrict__ x0,     // mode 1: [B*T][256]
    const float* __restrict__ x1,     // mode 1
    float* __restrict__ out,          // [2][B][256]
    int T, int B, int mode)
{
    __shared__ _Float16 xs[32 * 256];
    __shared__ _Float16 hs[32 * 256];

    const int tid   = threadIdx.x;
    const int lane  = tid & 31;
    const int w     = tid >> 5;      // wave id 0..7
    const int half  = lane >> 4;
    const int n16   = lane & 15;
    const int dir   = blockIdx.y;
    const int bbase = blockIdx.x * 32;

    // Preload per-lane gate biases (constant over time).
    float bi_r[2], bi_z[2], bi_n[2], bh_r[2], bh_z[2], bh_n[2];
#pragma unroll
    for (int i = 0; i < 2; ++i) {
        int j   = 32 * w + 16 * i + n16;
        bi_r[i] = bih[dir * 768 + j];
        bi_z[i] = bih[dir * 768 + 256 + j];
        bi_n[i] = bih[dir * 768 + 512 + j];
        bh_r[i] = bhh[dir * 768 + j];
        bh_z[i] = bhh[dir * 768 + 256 + j];
        bh_n[i] = bhh[dir * 768 + 512 + j];
    }

    float hold[2][2][8];  // [m-tile][col-subtile][row]
#pragma unroll
    for (int mt = 0; mt < 2; ++mt)
#pragma unroll
        for (int i = 0; i < 2; ++i)
#pragma unroll
            for (int r = 0; r < 8; ++r) hold[mt][i][r] = 0.f;

    for (int idx = tid; idx < 32 * 256; idx += 256) hs[idx] = (_Float16)0.f;
    // first in-loop barrier orders this init vs. WMMA reads

    for (int t = 0; t < T; ++t) {
        const int tt = dir ? (T - 1 - t) : t;

        // Launder the weight pointers through an empty asm so the compiler
        // cannot prove loop-invariance and hoist/spill the weight fragments;
        // they stream from L2 every step instead. Rebuild as address_space(1)
        // pointers so the loads lower to global_load_b128.
        unsigned long long wp_ih = (unsigned long long)wih_sw;
        unsigned long long wp_hh = (unsigned long long)whh_sw;
        asm volatile("" : "+s"(wp_ih), "+s"(wp_hh));
        gwp_t wih_p = (gwp_t)wp_ih;
        gwp_t whh_p = (gwp_t)wp_hh;

        // ---- stage x_t (f16) in LDS ----
        if (mode == 0) {
            for (int idx = tid; idx < 32 * 256; idx += 256) {
                int m = idx >> 8, e = idx & 255;
                int b = bbase + m;
                int tok = tokens[b * T + tt];
                xs[idx] = (_Float16)embed[(size_t)tok * 256 + e];
            }
        } else {
            for (int idx = tid; idx < 32 * 256; idx += 256) {
                int m = idx >> 8, e = idx & 255;
                int b = bbase + m;
                size_t o = (size_t)(b * T + tt) * 256 + e;
                xs[idx] = (_Float16)(x0[o] + x1[o]);
            }
        }
        __syncthreads();  // also orders previous step's hs writes

        // ---- WMMA: gx = x@Wih^T, gh = h@Whh^T  (kept separate) ----
        v8f accx[2][6], acch[2][6];
#pragma unroll
        for (int mt = 0; mt < 2; ++mt)
#pragma unroll
            for (int i = 0; i < 6; ++i)
#pragma unroll
                for (int r = 0; r < 8; ++r) {
                    accx[mt][i][r] = 0.f;
                    acch[mt][i][r] = 0.f;
                }

        for (int kt = 0; kt < 8; ++kt) {
            AF ax[2], ah[2];
#pragma unroll
            for (int mt = 0; mt < 2; ++mt)
#pragma unroll
                for (int v = 0; v < 8; ++v) {
                    // A layout: K = 2v + 16*half, row = lane&15
                    int k0 = kt * 32 + 2 * v + 16 * half;
                    int row = (mt * 16 + n16) * 256;
                    ax[mt].u[v] = *(const unsigned int*)&xs[row + k0];
                    ah[mt].u[v] = *(const unsigned int*)&hs[row + k0];
                }
#pragma unroll
            for (int i = 0; i < 6; ++i) {
                int nt = (i >> 1) * 16 + 2 * w + (i & 1);
                gwp_t px = wih_p + ((((size_t)dir * 48 + nt) * 8 + kt) * 32 + lane) * 8;
                gwp_t ph = whh_p + ((((size_t)dir * 48 + nt) * 8 + kt) * 32 + lane) * 8;
                AF bx, bh2;
#pragma unroll
                for (int v = 0; v < 8; ++v) { bx.u[v] = px[v]; bh2.u[v] = ph[v]; }
#pragma unroll
                for (int mt = 0; mt < 2; ++mt) {
                    accx[mt][i] = __builtin_amdgcn_wmma_f32_16x16x32_f16(
                        false, ax[mt].v, false, bx.v, (short)0, accx[mt][i],
                        false, false);
                    acch[mt][i] = __builtin_amdgcn_wmma_f32_16x16x32_f16(
                        false, ah[mt].v, false, bh2.v, (short)0, acch[mt][i],
                        false, false);
                }
            }
        }
        __syncthreads();  // all hs/xs reads done before hs is rewritten

        // ---- gates + h update: fully register-local per lane ----
#pragma unroll
        for (int mt = 0; mt < 2; ++mt)
#pragma unroll
            for (int i = 0; i < 2; ++i) {
                int j = 32 * w + 16 * i + n16;
#pragma unroll
                for (int r = 0; r < 8; ++r) {
                    int m = mt * 16 + 8 * half + r;  // C frag: VGPR r -> row
                    float gr = (accx[mt][0 + i][r] + bi_r[i]) +
                               (acch[mt][0 + i][r] + bh_r[i]);
                    float gz = (accx[mt][2 + i][r] + bi_z[i]) +
                               (acch[mt][2 + i][r] + bh_z[i]);
                    float rr = 1.f / (1.f + expf(-gr));
                    float zz = 1.f / (1.f + expf(-gz));
                    float nn = tanhf((accx[mt][4 + i][r] + bi_n[i]) +
                                     rr * (acch[mt][4 + i][r] + bh_n[i]));
                    float hn = (1.f - zz) * nn + zz * hold[mt][i][r];
                    hold[mt][i][r] = hn;
                    hs[m * 256 + j] = (_Float16)hn;
                }
            }
    }

    // ---- final hidden state ----
#pragma unroll
    for (int mt = 0; mt < 2; ++mt)
#pragma unroll
        for (int i = 0; i < 2; ++i) {
            int j = 32 * w + 16 * i + n16;
#pragma unroll
            for (int r = 0; r < 8; ++r) {
                int b = bbase + mt * 16 + 8 * half + r;
                out[((size_t)dir * B + b) * 256 + j] = hold[mt][i][r];
            }
        }
}

// ---------------------------------------------------------------------------
// Per-review head: p_batch = [rev0+rev1, normalize(uf)*uf_w]; r_stars.
// One block per review, 256 threads.
// ---------------------------------------------------------------------------
__global__ __launch_bounds__(256) void review_head_kernel(
    const float* __restrict__ rev0, const float* __restrict__ rev1,
    const float* __restrict__ user_feats, const float* __restrict__ uf_w,
    const float* __restrict__ w1, const float* __restrict__ b1,
    const float* __restrict__ w2, const float* __restrict__ b2,
    float* __restrict__ p_batch, float* __restrict__ r_stars)
{
    __shared__ float p[276];
    __shared__ float hid[128];
    __shared__ float red[64];
    __shared__ float nrm;
    const int r = blockIdx.x, tid = threadIdx.x;

    p[tid] = rev0[r * 256 + tid] + rev1[r * 256 + tid];
    if (tid == 0) {
        float s = 0.f;
        for (int u = 0; u < 20; ++u) { float v = user_feats[r * 20 + u]; s += v * v; }
        nrm = fmaxf(sqrtf(s), 1e-12f);
    }
    __syncthreads();
    if (tid < 20) p[256 + tid] = user_feats[r * 20 + tid] / nrm * uf_w[tid];
    __syncthreads();
    if (tid < 128) {
        float s = b1[tid];
        for (int k = 0; k < 276; ++k) s += p[k] * w1[tid * 276 + k];
        hid[tid] = selu_(s);
    }
    __syncthreads();
    if (tid < 64) red[tid] = hid[tid] * w2[tid] + hid[tid + 64] * w2[tid + 64];
    __syncthreads();
    for (int s = 32; s > 0; s >>= 1) {
        if (tid < s) red[tid] += red[tid + s];
        __syncthreads();
    }
    if (tid == 0) r_stars[r] = red[0] + b2[0];
    for (int i = tid; i < 276; i += 256) p_batch[r * 276 + i] = p[i];
}

// ---------------------------------------------------------------------------
// Review-level biGRU (batch 1, T=64, Din=276, H=256) + p_stars head.
// Single block, scalar VALU (26 MFLOP, not worth WMMA with M=1).
// ---------------------------------------------------------------------------
__global__ __launch_bounds__(256) void doc_gru_kernel(
    const float* __restrict__ p_batch,                 // [64][276]
    const float* __restrict__ wih,                     // [2][768][276]
    const float* __restrict__ whh,                     // [2][768][256]
    const float* __restrict__ bih, const float* __restrict__ bhh,
    const float* __restrict__ pw1, const float* __restrict__ pb1,
    const float* __restrict__ pw2, const float* __restrict__ pb2,
    float* __restrict__ p_stars)
{
    __shared__ float x[276];
    __shared__ float h[256];
    __shared__ float doc[256];
    __shared__ float hid[128];
    const int tid = threadIdx.x;
    const int j = tid;
    doc[tid] = 0.f;

    for (int dir = 0; dir < 2; ++dir) {
        h[tid] = 0.f;
        __syncthreads();
        const float* wi = wih + (size_t)dir * 768 * 276;
        const float* wh = whh + (size_t)dir * 768 * 256;
        for (int t = 0; t < 64; ++t) {
            int tt = dir ? 63 - t : t;
            for (int i = tid; i < 276; i += 256) x[i] = p_batch[tt * 276 + i];
            __syncthreads();
            float gxr = bih[dir * 768 + j];
            float gxz = bih[dir * 768 + 256 + j];
            float gxn = bih[dir * 768 + 512 + j];
            float ghr = bhh[dir * 768 + j];
            float ghz = bhh[dir * 768 + 256 + j];
            float ghn = bhh[dir * 768 + 512 + j];
            const float* wir = wi + (size_t)j * 276;
            const float* wiz = wi + (size_t)(256 + j) * 276;
            const float* win = wi + (size_t)(512 + j) * 276;
            for (int k = 0; k < 276; ++k) {
                float xv = x[k];
                gxr += xv * wir[k]; gxz += xv * wiz[k]; gxn += xv * win[k];
            }
            const float* whr = wh + (size_t)j * 256;
            const float* whz = wh + (size_t)(256 + j) * 256;
            const float* whn = wh + (size_t)(512 + j) * 256;
            for (int k = 0; k < 256; ++k) {
                float hv = h[k];
                ghr += hv * whr[k]; ghz += hv * whz[k]; ghn += hv * whn[k];
            }
            float rr = 1.f / (1.f + expf(-(gxr + ghr)));
            float zz = 1.f / (1.f + expf(-(gxz + ghz)));
            float nn = tanhf(gxn + rr * ghn);
            float hn = (1.f - zz) * nn + zz * h[j];
            __syncthreads();   // all reads of h done
            h[j] = hn;
            __syncthreads();
        }
        doc[tid] += h[tid];
        __syncthreads();
    }

    if (tid < 128) {
        float s = pb1[tid];
        for (int k = 0; k < 256; ++k) s += doc[k] * pw1[tid * 256 + k];
        hid[tid] = selu_(s);
    }
    __syncthreads();
    if (tid < 9) {
        float s = pb2[tid];
        for (int k = 0; k < 128; ++k) s += hid[k] * pw2[tid * 128 + k];
        p_stars[tid] = s;
    }
}

// ---------------------------------------------------------------------------
extern "C" void kernel_launch(void* const* d_in, const int* in_sizes, int n_in,
                              void* d_out, int out_size, void* d_ws, size_t ws_size,
                              hipStream_t stream)
{
    (void)in_sizes; (void)n_in; (void)out_size; (void)ws_size;

    const int*   inputs     = (const int*)  d_in[0];
    const float* user_feats = (const float*)d_in[3];
    const float* embed      = (const float*)d_in[4];
    const float* wg_wih = (const float*)d_in[5];
    const float* wg_whh = (const float*)d_in[6];
    const float* wg_bih = (const float*)d_in[7];
    const float* wg_bhh = (const float*)d_in[8];
    const float* sg_wih = (const float*)d_in[9];
    const float* sg_whh = (const float*)d_in[10];
    const float* sg_bih = (const float*)d_in[11];
    const float* sg_bhh = (const float*)d_in[12];
    const float* rg_wih = (const float*)d_in[13];
    const float* rg_whh = (const float*)d_in[14];
    const float* rg_bih = (const float*)d_in[15];
    const float* rg_bhh = (const float*)d_in[16];
    const float* rfc_w1 = (const float*)d_in[17];
    const float* rfc_b1 = (const float*)d_in[18];
    const float* rfc_w2 = (const float*)d_in[19];
    const float* rfc_b2 = (const float*)d_in[20];
    const float* pfc_w1 = (const float*)d_in[21];
    const float* pfc_b1 = (const float*)d_in[22];
    const float* pfc_w2 = (const float*)d_in[23];
    const float* pfc_b2 = (const float*)d_in[24];
    const float* uf_w   = (const float*)d_in[25];

    // workspace layout (all 256B-aligned by construction)
    char* ws = (char*)d_ws;
    const size_t SW = (size_t)2 * 48 * 8 * 32 * 8;  // 196608 dwords per matrix
    unsigned int* wg_ih_sw = (unsigned int*)ws; ws += SW * 4;
    unsigned int* wg_hh_sw = (unsigned int*)ws; ws += SW * 4;
    unsigned int* sg_ih_sw = (unsigned int*)ws; ws += SW * 4;
    unsigned int* sg_hh_sw = (unsigned int*)ws; ws += SW * 4;
    float* out_word = (float*)ws; ws += (size_t)2 * 2048 * 256 * 4;  // [2][2048][256]
    float* out_sent = (float*)ws; ws += (size_t)2 * 64 * 256 * 4;    // [2][64][256]
    float* p_batch  = (float*)ws; ws += (size_t)64 * 276 * 4;        // [64][276]

    float* p_stars = (float*)d_out;      // [9]
    float* r_stars = (float*)d_out + 9;  // [64]

    dim3 blk(256);
    dim3 swz((unsigned)((SW + 255) / 256));
    swizzle_weights_kernel<<<swz, blk, 0, stream>>>(wg_wih, wg_ih_sw, (int)SW);
    swizzle_weights_kernel<<<swz, blk, 0, stream>>>(wg_whh, wg_hh_sw, (int)SW);
    swizzle_weights_kernel<<<swz, blk, 0, stream>>>(sg_wih, sg_ih_sw, (int)SW);
    swizzle_weights_kernel<<<swz, blk, 0, stream>>>(sg_whh, sg_hh_sw, (int)SW);

    // word-level biGRU: 2048 sentences (32 per block), T=64, token-gather input
    bigru16_wmma_kernel<<<dim3(64, 2), blk, 0, stream>>>(
        wg_ih_sw, wg_hh_sw, wg_bih, wg_bhh,
        inputs, embed, nullptr, nullptr,
        out_word, 64, 2048, 0);

    // sentence-level biGRU: 64 reviews (32 per block), T=32,
    // input = sum of word-dir outputs
    bigru16_wmma_kernel<<<dim3(2, 2), blk, 0, stream>>>(
        sg_ih_sw, sg_hh_sw, sg_bih, sg_bhh,
        nullptr, nullptr, out_word, out_word + (size_t)2048 * 256,
        out_sent, 32, 64, 1);

    // p_batch + r_stars
    review_head_kernel<<<dim3(64), blk, 0, stream>>>(
        out_sent, out_sent + (size_t)64 * 256,
        user_feats, uf_w, rfc_w1, rfc_b1, rfc_w2, rfc_b2,
        p_batch, r_stars);

    // review-level biGRU + p_stars
    doc_gru_kernel<<<dim3(1), blk, 0, stream>>>(
        p_batch, rg_wih, rg_whh, rg_bih, rg_bhh,
        pfc_w1, pfc_b1, pfc_w2, pfc_b2, p_stars);
}
